// MultiScaleChebConv_23605140258865
// MI455X (gfx1250) — matrix operand order repaired and checked by reading, not compiled, not run on hardware
//
#include <hip/hip_runtime.h>

#define NN   10000
#define EE   160000
#define FIN  1152
#define FOUT 1152   // 4 * 288

typedef _Float16 v16h __attribute__((ext_vector_type(16)));
typedef _Float16 v8h  __attribute__((ext_vector_type(8)));
typedef _Float16 v4h  __attribute__((ext_vector_type(4)));
typedef float    v8f  __attribute__((ext_vector_type(8)));

// ---------- graph normalization ----------

__global__ void k_initT(float* __restrict__ dst, const float* __restrict__ src,
                        float coef, long n4) {
    long t = blockIdx.x * (long)blockDim.x + threadIdx.x;
    if (t >= n4) return;
    float4 v;
    if (coef == 0.0f) {
        v = make_float4(0.f, 0.f, 0.f, 0.f);
    } else {
        float4 s = ((const float4*)src)[t];
        v = make_float4(coef * s.x, coef * s.y, coef * s.z, coef * s.w);
    }
    ((float4*)dst)[t] = v;
}

__global__ void k_degree(const int* __restrict__ ei, float* __restrict__ deg) {
    int e = blockIdx.x * blockDim.x + threadIdx.x;
    if (e >= EE) return;
    unsafeAtomicAdd(&deg[ei[e]], 1.0f);
}

__global__ void k_dinv(float* __restrict__ deg) {
    int i = blockIdx.x * blockDim.x + threadIdx.x;
    if (i >= NN) return;
    float v = deg[i];
    deg[i] = (v > 0.0f) ? rsqrtf(v) : 0.0f;
}

__global__ void k_norm(const int* __restrict__ ei, const float* __restrict__ dinv,
                       float* __restrict__ nrm) {
    int e = blockIdx.x * blockDim.x + threadIdx.x;
    if (e >= EE) return;
    nrm[e] = -dinv[ei[e]] * dinv[ei[EE + e]];
}

// scatter: dst[col] += scale * norm[e] * h[row], 4 floats per thread
__global__ void k_scatter(const int* __restrict__ ei, const float* __restrict__ nrm,
                          const float* __restrict__ h, float* __restrict__ dst,
                          float scale) {
    long t = blockIdx.x * (long)blockDim.x + threadIdx.x;
    const long total = (long)EE * (FIN / 4);
    if (t >= total) return;
    int e = (int)(t / (FIN / 4));
    int c = (int)(t % (FIN / 4));
    int r   = ei[e];
    int col = ei[EE + e];
    float w = scale * nrm[e];
    float4 v = ((const float4*)(h + (long)r * FIN))[c];
    float* d = dst + (long)col * FIN + 4 * c;
    unsafeAtomicAdd(d + 0, w * v.x);
    unsafeAtomicAdd(d + 1, w * v.y);
    unsafeAtomicAdd(d + 2, w * v.z);
    unsafeAtomicAdd(d + 3, w * v.w);
}

__global__ void k_f32tof16(const float* __restrict__ src, _Float16* __restrict__ dst,
                           long n4) {
    long t = blockIdx.x * (long)blockDim.x + threadIdx.x;
    if (t >= n4) return;
    float4 v = ((const float4*)src)[t];
    v4h o;
    o.x = (_Float16)v.x; o.y = (_Float16)v.y; o.z = (_Float16)v.z; o.w = (_Float16)v.w;
    ((v4h*)dst)[t] = o;
}

// ---------- weight packing ----------

__device__ __forceinline__ int d_startcol(int k) {
    int s = k - 3; return (s > 0) ? 288 * s : 0;
}
__device__ __forceinline__ long d_colbase(int k) {
    long s = 0;
    for (int j = 0; j < k; ++j) s += FIN - ((j > 3) ? 288 * (j - 3) : 0);
    return s;
}

// Pack W_c[k][f][o] (fp32, [Kc,1152,288]) into transposed f16 Wp[k-block][packed_col][f]
__global__ void k_pack(const float* __restrict__ W, _Float16* __restrict__ Wp,
                       int Kc, int c) {
    const long per_k = 288L * FIN;
    long t = blockIdx.x * (long)blockDim.x + threadIdx.x;
    if (t >= (long)Kc * per_k) return;
    int  k = (int)(t / per_k);
    long r = t - (long)k * per_k;
    int  o = (int)(r / FIN);
    int  f = (int)(r % FIN);
    long pcol = (long)c * 288 - d_startcol(k) + o;   // column within pack k
    Wp[(d_colbase(k) + pcol) * FIN + f] = (_Float16)W[((long)k * FIN + f) * 288 + o];
}

__global__ void k_bias(const float* __restrict__ b0, const float* __restrict__ b1,
                       const float* __restrict__ b2, const float* __restrict__ b3,
                       float* __restrict__ out) {
    long t = blockIdx.x * (long)blockDim.x + threadIdx.x;
    if (t >= (long)NN * FOUT) return;
    int o = (int)(t % FOUT);
    float v;
    if      (o < 288) v = b0[o];
    else if (o < 576) v = b1[o - 288];
    else if (o < 864) v = b2[o - 576];
    else              v = b3[o - 864];
    out[t] = v;
}

// ---------- WMMA GEMM: out[:, startcol + n] += A(f16)[m,f] * Bt(f16)[n,f] ----------
// A: [10000, 1152] row-major f16.  Bt: [width, 1152] row-major f16 (weights transposed).
// One wave per block; wave tile = 16 rows x 48 cols (3 wmma chains, NO guards:
// every per-k width (1152/864/576/288) is a multiple of 48, and 10000 = 625*16).

__global__ void k_gemm(const _Float16* __restrict__ A, const _Float16* __restrict__ Bt,
                       float* __restrict__ out, int startcol) {
    const int lane = threadIdx.x;
    const int lr = lane & 15;
    const int hi = lane >> 4;
    const int m0 = blockIdx.x * 16;
    const int n0 = blockIdx.y * 48;

    v8f acc0 = 0, acc1 = 0, acc2 = 0;

    // A fragment (16x32 f16): lanes 0-15 hold M=lr, K = kb+0..15;
    // lanes 16-31 hold M=lr, K = kb+16..31.
    const _Float16* arow = A + (long)(m0 + lr) * FIN + hi * 16;
    // B fragment (32x16 f16): lanes 0-15 hold K=kb+0..15 of col n0+lr,
    // lanes 16-31 hold K=kb+16..31; contiguous in the transposed Bt layout.
    const _Float16* brow0 = Bt + (long)(n0 + lr) * FIN + hi * 16;
    const _Float16* brow1 = brow0 + 16L * FIN;
    const _Float16* brow2 = brow1 + 16L * FIN;

    for (int kb = 0; kb < FIN; kb += 32) {
        v16h af = __builtin_shufflevector(*(const v8h*)(arow + kb),
                                          *(const v8h*)(arow + kb + 8),
                                          0,1,2,3,4,5,6,7,8,9,10,11,12,13,14,15);
        v16h b0 = __builtin_shufflevector(*(const v8h*)(brow0 + kb),
                                          *(const v8h*)(brow0 + kb + 8),
                                          0,1,2,3,4,5,6,7,8,9,10,11,12,13,14,15);
        v16h b1 = __builtin_shufflevector(*(const v8h*)(brow1 + kb),
                                          *(const v8h*)(brow1 + kb + 8),
                                          0,1,2,3,4,5,6,7,8,9,10,11,12,13,14,15);
        v16h b2 = __builtin_shufflevector(*(const v8h*)(brow2 + kb),
                                          *(const v8h*)(brow2 + kb + 8),
                                          0,1,2,3,4,5,6,7,8,9,10,11,12,13,14,15);
        acc0 = __builtin_amdgcn_wmma_f32_16x16x32_f16(false, af, false, b0,
                                                      (short)0, acc0, false, false);
        acc1 = __builtin_amdgcn_wmma_f32_16x16x32_f16(false, af, false, b1,
                                                      (short)0, acc1, false, false);
        acc2 = __builtin_amdgcn_wmma_f32_16x16x32_f16(false, af, false, b2,
                                                      (short)0, acc2, false, false);
    }

    // C layout: VGPR r, lanes 0-15 -> (M=r, N=lane); lanes 16-31 -> (M=r+8, N=lane-16)
    float* obase = out + (long)(m0 + 8 * hi) * FOUT + startcol + n0 + lr;
#pragma unroll
    for (int r = 0; r < 8; ++r) {
        float* orow = obase + (long)r * FOUT;
        orow[0]  += acc0[r];
        orow[16] += acc1[r];
        orow[32] += acc2[r];
    }
}

// ---------- host ----------

static inline int h_startcol(int k) { int s = k - 3; return (s > 0) ? 288 * s : 0; }
static inline long h_colbase(int k) {
    long s = 0;
    for (int j = 0; j < k; ++j) s += FIN - ((j > 3) ? 288 * (j - 3) : 0);
    return s;
}

extern "C" void kernel_launch(void* const* d_in, const int* in_sizes, int n_in,
                              void* d_out, int out_size, void* d_ws, size_t ws_size,
                              hipStream_t stream) {
    (void)in_sizes; (void)n_in; (void)out_size; (void)ws_size;

    const float* x  = (const float*)d_in[0];
    const int*   ei = (const int*)d_in[1];
    const float* W[4]  = {(const float*)d_in[2], (const float*)d_in[4],
                          (const float*)d_in[6], (const float*)d_in[8]};
    const float* bb[4] = {(const float*)d_in[3], (const float*)d_in[5],
                          (const float*)d_in[7], (const float*)d_in[9]};
    float* out = (float*)d_out;

    char* p = (char*)d_ws;
    float*    deg  = (float*)p;     p += 40192;                    // N f32 (rounded)
    float*    nrm  = (float*)p;     p += (size_t)EE * 4;           // E f32
    float*    bufA = (float*)p;     p += (size_t)NN * FIN * 4;     // T ping
    float*    bufB = (float*)p;     p += (size_t)NN * FIN * 4;     // T pong
    _Float16* Tf16 = (_Float16*)p;  p += (size_t)NN * FIN * 2;     // f16 copy of T_k
    _Float16* Wp   = (_Float16*)p;                                  // packed weights (6336*1152 f16)

    const long nTf4 = (long)NN * FIN / 4;
    const int  TB = 256;
    auto nb = [](long n) { return (int)((n + 255) / 256); };

    // normalization: deg -> dinv -> norm
    k_initT<<<nb(2500), TB, 0, stream>>>(deg, deg, 0.0f, 2500);
    k_degree<<<nb(EE), TB, 0, stream>>>(ei, deg);
    k_dinv<<<nb(NN), TB, 0, stream>>>(deg);
    k_norm<<<nb(EE), TB, 0, stream>>>(ei, deg, nrm);

    // pack weights (transposed f16) and init output with bias
    for (int c = 0; c < 4; ++c) {
        int Kc = 4 + c;
        k_pack<<<nb((long)Kc * 288 * FIN), TB, 0, stream>>>(W[c], Wp, Kc, c);
    }
    k_bias<<<nb((long)NN * FOUT), TB, 0, stream>>>(bb[0], bb[1], bb[2], bb[3], out);

    // k = 0: T0 = x
    k_f32tof16<<<nb(nTf4), TB, 0, stream>>>(x, Tf16, nTf4);
    {
        dim3 g(NN / 16, FIN / 48);
        k_gemm<<<g, 32, 0, stream>>>(Tf16, Wp, out, 0);
    }

    // k = 1..6: T_k = 2*prop(T_{k-1}) - T_{k-2}  (k=1: T_1 = prop(T_0))
    for (int k = 1; k <= 6; ++k) {
        float*       dst = (k & 1) ? bufA : bufB;                       // T_k
        const float* h   = (k == 1) ? x : ((k & 1) ? bufB : bufA);      // T_{k-1}
        const float* p2  = (k <= 2) ? x : dst;                          // T_{k-2}
        float coef  = (k == 1) ? 0.0f : -1.0f;
        float scale = (k == 1) ? 1.0f : 2.0f;

        k_initT<<<nb(nTf4), TB, 0, stream>>>(dst, p2, coef, nTf4);
        k_scatter<<<nb((long)EE * (FIN / 4)), TB, 0, stream>>>(ei, nrm, h, dst, scale);
        k_f32tof16<<<nb(nTf4), TB, 0, stream>>>(dst, Tf16, nTf4);

        int sc = h_startcol(k), wd = FIN - sc;
        dim3 g(NN / 16, wd / 48);
        k_gemm<<<g, 32, 0, stream>>>(Tf16, Wp + h_colbase(k) * FIN, out, sc);
    }
}
